// GroupAttention_71356586656140
// MI455X (gfx1250) — matrix operand order repaired and checked by reading, not compiled, not run on hardware
//
#include <hip/hip_runtime.h>
#include <hip/hip_bf16.h>

// ---------------------------------------------------------------------------
// Fused grouped attention for MI455X (gfx1250, wave32).
//   qkv = x @ w_qkv            (bf16 WMMA, f32 accum)
//   attn within 4-token groups (VALU, tiny)
//   out = o @ w_proj + b       (bf16 WMMA, f32 accum)
// ---------------------------------------------------------------------------

typedef __attribute__((ext_vector_type(16))) __bf16 v16bf;
typedef __attribute__((ext_vector_type(8)))  float  v8f;

#define ROWS   32768   // B*N = 4*8192
#define C_DIM  1024
#define H_NUM  16
#define DH     64
#define GRP    4
#define TILE_M 32      // tokens per workgroup (multiple of GRP, divides 8192)
#define KSTEP  32      // K per WMMA

union FragU { v16bf v; uint4 u[2]; };

__device__ __forceinline__ unsigned short f2bf(float f) {
  union { float f; unsigned u; } x; x.f = f;
  unsigned u = x.u;
  u += 0x7fffu + ((u >> 16) & 1u);   // round-to-nearest-even
  return (unsigned short)(u >> 16);
}

// A fragment (16x32 bf16) from a row-major bf16 buffer, leading dim ld.
// Contiguous-k layout used here (same as B): lane<16 -> M=lane,   K = kk+{0..15}
//             lane>=16 -> M=lane-16, K = kk+{16..31}
__device__ __forceinline__ v16bf load_frag_a(const unsigned short* base, int ld,
                                             int m0, int kk, int lane) {
  const int m  = m0 + (lane & 15);
  const int kb = kk + ((lane >> 4) << 4);
  FragU r;
  r.u[0] = *(const uint4*)(base + m * ld + kb);
  r.u[1] = *(const uint4*)(base + m * ld + kb + 8);
  return r.v;
}

// B fragment (32x16 bf16) from column-major (pre-transposed) weights wT[n][k].
// Contiguous-k layout used here (same as A): N = lane&15, K = kk + 16*(lane>=16) + e -> 16 contiguous K / lane
__device__ __forceinline__ v16bf load_frag_b(const unsigned short* wT, int ld,
                                             int n0, int kk, int lane) {
  const int n  = n0 + (lane & 15);
  const int kb = kk + ((lane >> 4) << 4);
  const uint4* p = (const uint4*)(wT + (size_t)n * ld + kb);
  FragU r;
  r.u[0] = p[0];
  r.u[1] = p[1];
  return r.v;
}

// C/D layout: VGPR r -> (M = m0 + r + 8*(lane>=16), N = n0 + lane&15)
__device__ __forceinline__ void store_c_lds(float* dst, int ld, int m0, int n0,
                                            v8f acc, int lane) {
  const int n  = n0 + (lane & 15);
  const int mb = m0 + ((lane >> 4) << 3);
#pragma unroll
  for (int r = 0; r < 8; ++r) dst[(mb + r) * ld + n] = acc[r];
}

// ---------------------------------------------------------------------------
// Prep: transpose + convert weights fp32 -> bf16 (one-time, L2-resident after)
// ---------------------------------------------------------------------------
__global__ void prep_weights(const float* __restrict__ w_qkv,
                             const float* __restrict__ w_proj,
                             unsigned short* __restrict__ wqkvT,
                             unsigned short* __restrict__ wprojT) {
  const size_t NQKV = (size_t)3 * C_DIM * C_DIM;
  const size_t NPRJ = (size_t)C_DIM * C_DIM;
  size_t i = (size_t)blockIdx.x * blockDim.x + threadIdx.x;
  if (i < NQKV) {
    int n = (int)(i / C_DIM), k = (int)(i % C_DIM);
    wqkvT[i] = f2bf(w_qkv[(size_t)k * (3 * C_DIM) + n]);
  } else if (i < NQKV + NPRJ) {
    size_t j = i - NQKV;
    int n = (int)(j / C_DIM), k = (int)(j % C_DIM);
    wprojT[j] = f2bf(w_proj[(size_t)k * C_DIM + n]);
  }
}

// ---------------------------------------------------------------------------
// Fused main kernel: one workgroup owns 32 tokens (8 groups of 4).
// Dynamic LDS: x tile (64KB bf16) | o tile (64KB bf16) | qkv scratch (24KB f32)
// ---------------------------------------------------------------------------
__global__ __launch_bounds__(256)
void fused_group_attn(const float* __restrict__ x,
                      const unsigned short* __restrict__ wqkvT,
                      const unsigned short* __restrict__ wprojT,
                      const float* __restrict__ bias,
                      float* __restrict__ out) {
  extern __shared__ unsigned char smem[];
  unsigned short* x_s = (unsigned short*)smem;            // [TILE_M][C_DIM]
  unsigned short* o_s = x_s + TILE_M * C_DIM;             // [TILE_M][C_DIM]
  float*          s_s = (float*)(o_s + TILE_M * C_DIM);   // [TILE_M][192] q|k|v
  __shared__ float attn_s[TILE_M / GRP][GRP][GRP];

  const int tid  = threadIdx.x;
  const int lane = tid & 31;
  const int wave = tid >> 5;
  const int row0 = blockIdx.x * TILE_M;
  const float scale = 0.125f;   // Dh^-0.5

  // ---- stage x tile -> LDS (fp32 -> bf16), fully coalesced b128 loads
  {
    const float4* xg = (const float4*)(x + (size_t)row0 * C_DIM);
#pragma unroll
    for (int e = 0; e < 32; ++e) {
      int lin = e * 256 + tid;              // 8192 float4 per tile
      float4 v = xg[lin];
      uint2 p;
      p.x = (unsigned)f2bf(v.x) | ((unsigned)f2bf(v.y) << 16);
      p.y = (unsigned)f2bf(v.z) | ((unsigned)f2bf(v.w) << 16);
      *(uint2*)(x_s + lin * 4) = p;
    }
  }
  __syncthreads();

  // =======================================================================
  // Phase A: per head, qkv tile GEMM (WMMA) + 4-wide group attention (VALU)
  // =======================================================================
  for (int h = 0; h < H_NUM; ++h) {
    // ---- GEMM: s[32][192] = x_tile @ wqkv[:, {q,k,v} cols of head h]
    {
      const int t0 = wave * 3;            // 24 tiles = 2 row-tiles x 12 col-tiles
      const int rt = t0 / 12;             // identical for all 3 tiles of a wave
      const int m0 = rt * 16;
      const int ct0 = (t0 + 0) % 12, ct1 = (t0 + 1) % 12, ct2 = (t0 + 2) % 12;
      const int n0 = (ct0 >> 2) * C_DIM + h * DH + (ct0 & 3) * 16;
      const int n1 = (ct1 >> 2) * C_DIM + h * DH + (ct1 & 3) * 16;
      const int n2 = (ct2 >> 2) * C_DIM + h * DH + (ct2 & 3) * 16;
      v8f acc0 = {}, acc1 = {}, acc2 = {};
      for (int kk = 0; kk < C_DIM; kk += KSTEP) {
        v16bf a = load_frag_a(x_s, C_DIM, m0, kk, lane);
        __builtin_prefetch(wqkvT + (size_t)(n0 + (lane & 15)) * C_DIM + kk + KSTEP, 0, 0);
        v16bf b0 = load_frag_b(wqkvT, C_DIM, n0, kk, lane);
        acc0 = __builtin_amdgcn_wmma_f32_16x16x32_bf16(false, a, false, b0,
                                                       (short)0, acc0, false, false);
        v16bf b1 = load_frag_b(wqkvT, C_DIM, n1, kk, lane);
        acc1 = __builtin_amdgcn_wmma_f32_16x16x32_bf16(false, a, false, b1,
                                                       (short)0, acc1, false, false);
        v16bf b2 = load_frag_b(wqkvT, C_DIM, n2, kk, lane);
        acc2 = __builtin_amdgcn_wmma_f32_16x16x32_bf16(false, a, false, b2,
                                                       (short)0, acc2, false, false);
      }
      store_c_lds(s_s, 192, m0, ct0 * 16, acc0, lane);
      store_c_lds(s_s, 192, m0, ct1 * 16, acc1, lane);
      store_c_lds(s_s, 192, m0, ct2 * 16, acc2, lane);
    }
    __syncthreads();

    // ---- scores: 8 groups x 4x4 pairs; s_s cols [0,64)=q [64,128)=k [128,192)=v
    if (tid < 128) {
      const int g = tid >> 4, i = (tid >> 2) & 3, j = tid & 3;
      const float* qrow = s_s + (g * GRP + i) * 192;
      const float* krow = s_s + (g * GRP + j) * 192 + 64;
      float acc = 0.f;
#pragma unroll 8
      for (int d = 0; d < DH; ++d) acc += qrow[d] * krow[d];
      attn_s[g][i][j] = acc * scale;
    }
    __syncthreads();

    // ---- softmax over each 4-row (one thread per (group, query))
    if (tid < TILE_M) {
      const int g = tid >> 2, i = tid & 3;
      float a0 = attn_s[g][i][0], a1 = attn_s[g][i][1];
      float a2 = attn_s[g][i][2], a3 = attn_s[g][i][3];
      float m = fmaxf(fmaxf(a0, a1), fmaxf(a2, a3));
      float e0 = __expf(a0 - m), e1 = __expf(a1 - m);
      float e2 = __expf(a2 - m), e3 = __expf(a3 - m);
      float inv = 1.0f / (e0 + e1 + e2 + e3);
      attn_s[g][i][0] = e0 * inv; attn_s[g][i][1] = e1 * inv;
      attn_s[g][i][2] = e2 * inv; attn_s[g][i][3] = e3 * inv;
    }
    __syncthreads();

    // ---- o[tok][h*64 + d] = sum_j p[g][i][j] * v[g*4+j][d]  -> bf16 LDS
#pragma unroll
    for (int e = 0; e < 8; ++e) {
      int lin = e * 256 + tid;            // 32 tokens * 64 dims
      int tok = lin >> 6, d = lin & 63;
      int g = tok >> 2, i = tok & 3;
      const float* vb = s_s + (g * GRP) * 192 + 128 + d;
      float o = attn_s[g][i][0] * vb[0]
              + attn_s[g][i][1] * vb[192]
              + attn_s[g][i][2] * vb[2 * 192]
              + attn_s[g][i][3] * vb[3 * 192];
      o_s[tok * C_DIM + h * DH + d] = f2bf(o);
    }
    __syncthreads();   // s_s is rewritten by the next head's GEMM
  }

  // =======================================================================
  // Phase B: out_tile(32x1024) = o_tile @ w_proj + bias   (WMMA)
  // 128 tiles: wave w owns ct = (w&3)*16 + j, rt = w>>2; 4 passes of 4 tiles
  // =======================================================================
  {
    const int m0 = (wave >> 2) * 16;
    for (int pass = 0; pass < 4; ++pass) {
      const int ctp = (wave & 3) * 16 + pass * 4;
      const int n0 = (ctp + 0) * 16, n1 = (ctp + 1) * 16;
      const int n2 = (ctp + 2) * 16, n3 = (ctp + 3) * 16;
      const int nc = lane & 15;
      const float bv0 = bias[n0 + nc], bv1 = bias[n1 + nc];
      const float bv2 = bias[n2 + nc], bv3 = bias[n3 + nc];
      v8f acc0, acc1, acc2, acc3;
#pragma unroll
      for (int r = 0; r < 8; ++r) { acc0[r] = bv0; acc1[r] = bv1; acc2[r] = bv2; acc3[r] = bv3; }

      for (int kk = 0; kk < C_DIM; kk += KSTEP) {
        v16bf a = load_frag_a(o_s, C_DIM, m0, kk, lane);
        __builtin_prefetch(wprojT + (size_t)(n0 + nc) * C_DIM + kk + KSTEP, 0, 0);
        v16bf b = load_frag_b(wprojT, C_DIM, n0, kk, lane);
        acc0 = __builtin_amdgcn_wmma_f32_16x16x32_bf16(false, a, false, b,
                                                       (short)0, acc0, false, false);
        b = load_frag_b(wprojT, C_DIM, n1, kk, lane);
        acc1 = __builtin_amdgcn_wmma_f32_16x16x32_bf16(false, a, false, b,
                                                       (short)0, acc1, false, false);
        b = load_frag_b(wprojT, C_DIM, n2, kk, lane);
        acc2 = __builtin_amdgcn_wmma_f32_16x16x32_bf16(false, a, false, b,
                                                       (short)0, acc2, false, false);
        b = load_frag_b(wprojT, C_DIM, n3, kk, lane);
        acc3 = __builtin_amdgcn_wmma_f32_16x16x32_bf16(false, a, false, b,
                                                       (short)0, acc3, false, false);
      }

      const int mb = m0 + ((lane >> 4) << 3);
#pragma unroll
      for (int r = 0; r < 8; ++r) {
        const size_t rowg = (size_t)(row0 + mb + r) * C_DIM;
        out[rowg + n0 + nc] = acc0[r];
        out[rowg + n1 + nc] = acc1[r];
        out[rowg + n2 + nc] = acc2[r];
        out[rowg + n3 + nc] = acc3[r];
      }
    }
  }
}

// ---------------------------------------------------------------------------
extern "C" void kernel_launch(void* const* d_in, const int* in_sizes, int n_in,
                              void* d_out, int out_size, void* d_ws, size_t ws_size,
                              hipStream_t stream) {
  (void)in_sizes; (void)n_in; (void)out_size; (void)ws_size;
  const float* x      = (const float*)d_in[0];
  const float* w_qkv  = (const float*)d_in[1];
  const float* w_proj = (const float*)d_in[2];
  const float* b_proj = (const float*)d_in[3];
  // d_in[4] = causal flag; the reference never applies it.

  // workspace: bf16 transposed weights (8 MB total)
  unsigned short* wqkvT  = (unsigned short*)d_ws;                  // [3072][1024]
  unsigned short* wprojT = wqkvT + (size_t)3 * C_DIM * C_DIM;      // [1024][1024]

  {
    const size_t total = (size_t)4 * C_DIM * C_DIM;                // 4M elements
    const int threads = 256;
    const int blocks = (int)((total + threads - 1) / threads);
    prep_weights<<<blocks, threads, 0, stream>>>(w_qkv, w_proj, wqkvT, wprojT);
  }
  {
    dim3 grid(ROWS / TILE_M);                                      // 1024 blocks
    const size_t shmem = (size_t)TILE_M * C_DIM * 2 * 2            // x_s + o_s
                       + (size_t)TILE_M * 192 * 4;                 // s_s
    fused_group_attn<<<grid, 256, shmem, stream>>>(x, wqkvT, wprojT, b_proj,
                                                   (float*)d_out);
  }
}
